// KascadeReuseAttention_51642686767695
// MI455X (gfx1250) — matrix operand order repaired and checked by reading, compiled, not run on hardware
//
#include <hip/hip_runtime.h>

typedef __attribute__((ext_vector_type(16))) __bf16 v16bf;
typedef __attribute__((ext_vector_type(8)))  __bf16 v8bf;
typedef __attribute__((ext_vector_type(8)))  float  v8f;

#define S_    4096
#define E_    2048
#define H_    16
#define D_    128
#define T_    32
#define KANC_ 8
#define TILE_ 128
#define NEGV  (-1.0e10f)
#define SCALE_ 0.08838834764831845f   // 1/sqrt(128)

// ---- CDNA5 async memory->LDS helpers ---------------------------------------

__device__ __forceinline__ unsigned lds_off(const void* p) {
  // generic pointers to LDS: low 32 bits are the wave-logical LDS byte address
  return (unsigned)(size_t)p;
}

__device__ __forceinline__ void async_copy16(unsigned ldsAddr, const void* gaddr) {
  asm volatile("global_load_async_to_lds_b128 %0, %1, off"
               :: "v"(ldsAddr), "v"(gaddr) : "memory");
}

#define WAIT_ASYNC(lit) asm volatile("s_wait_asynccnt " lit ::: "memory")

// ---- WMMA helpers -----------------------------------------------------------

__device__ __forceinline__ v16bf mk16(const __bf16* lo, const __bf16* hi) {
  v8bf a = *(const v8bf*)lo;
  v8bf b = *(const v8bf*)hi;
  v16bf r;
#pragma unroll
  for (int i = 0; i < 8; i++) { r[i] = a[i]; r[i + 8] = b[i]; }
  return r;
}

__device__ __forceinline__ v8f wmma_bf16(v16bf a, v16bf b, v8f c) {
  return __builtin_amdgcn_wmma_f32_16x16x32_bf16(false, a, false, b,
                                                 (short)0, c, false, false);
}

// ---- fp32 -> bf16 conversion (row-major) ------------------------------------

__global__ __launch_bounds__(256) void cvt_bf16_kernel(const float* __restrict__ src,
                                                       __bf16* __restrict__ dst, int n) {
  int i = blockIdx.x * 256 + threadIdx.x;
  if (i < n) dst[i] = (__bf16)src[i];
}

// ---- fp32 [K][N] -> bf16 transposed [N][K] (one-time weight repack) ---------

__global__ __launch_bounds__(256) void cvt_bf16_t_kernel(const float* __restrict__ src,
                                                         __bf16* __restrict__ dst) {
  int gid = blockIdx.x * 256 + threadIdx.x;   // over E_*E_
  int k = gid >> 11;                          // E_ = 2048
  int n = gid & (E_ - 1);
  dst[(size_t)n * E_ + k] = (__bf16)src[gid];
}

// ---- bf16 GEMM: C[M,N] = A[M,K] * Bt[N,K]^T, f32 accumulate -----------------
// 256 threads (8 wave32, 4 along M x 2 along N), 128x128 tile, double-buffered
// async LDS staging (4 async_to_lds_b128 per thread per k-slice).

__global__ __launch_bounds__(256) void gemm_bf16_kernel(const __bf16* __restrict__ A,
                                                        const __bf16* __restrict__ Bt,
                                                        float* __restrict__ C,
                                                        int M, int N, int Kd) {
  __shared__ __bf16 sA[2][128 * 32];   // [row][k]
  __shared__ __bf16 sB[2][128 * 32];   // [n][k] (Bt already transposed)
  const int tid  = threadIdx.x;
  const int lane = tid & 31;
  const int w    = tid >> 5;
  const int hf   = lane >> 4;
  const int l16  = lane & 15;
  const int wm   = w & 3;
  const int wn   = w >> 2;
  const int mBase = blockIdx.y * 128;
  const int nBase = blockIdx.x * 128;

  const v8f vz = {};
  v8f acc[2][4];
#pragma unroll
  for (int a = 0; a < 2; a++)
#pragma unroll
    for (int b = 0; b < 4; b++) acc[a][b] = vz;

  // stage one 128x32 A slice + 128x32 Bt slice into buffer `bi`
  auto stage = [&](int bi, int k0) {
#pragma unroll
    for (int i = 0; i < 2; i++) {
      int off = (tid + i * 256) * 8;          // 0..4095, step 8
      int row = off >> 5, kk = off & 31;
      async_copy16(lds_off(&sA[bi][off]), &A [(size_t)(mBase + row) * Kd + k0 + kk]);
      async_copy16(lds_off(&sB[bi][off]), &Bt[(size_t)(nBase + row) * Kd + k0 + kk]);
    }
  };

  stage(0, 0);
  int buf = 0;
  for (int k0 = 0; k0 < Kd; k0 += 32, buf ^= 1) {
    const bool more = (k0 + 32) < Kd;
    if (more) {
      stage(buf ^ 1, k0 + 32);   // prefetch next slice into other buffer
      WAIT_ASYNC("0x4");         // drain previous slice (4 newest stay in flight)
    } else {
      WAIT_ASYNC("0x0");
    }
    __syncthreads();             // all waves' portions of buf landed

    const __bf16* cA = sA[buf];
    const __bf16* cB = sB[buf];
    v16bf af[2], bf4[4];
#pragma unroll
    for (int mi = 0; mi < 2; mi++) {
      int row = wm * 32 + mi * 16 + l16;
      af[mi] = mk16(&cA[row * 32 + hf * 8], &cA[row * 32 + 16 + hf * 8]);
    }
#pragma unroll
    for (int ni = 0; ni < 4; ni++) {
      int col = wn * 64 + ni * 16 + l16;
      bf4[ni] = mk16(&cB[col * 32 + hf * 8], &cB[col * 32 + 16 + hf * 8]);
    }
#pragma unroll
    for (int mi = 0; mi < 2; mi++)
#pragma unroll
      for (int ni = 0; ni < 4; ni++)
        acc[mi][ni] = wmma_bf16(af[mi], bf4[ni], acc[mi][ni]);

    __syncthreads();             // done reading buf before it is overwritten
  }

#pragma unroll
  for (int mi = 0; mi < 2; mi++)
#pragma unroll
    for (int ni = 0; ni < 4; ni++)
#pragma unroll
      for (int r = 0; r < 8; r++) {
        int m = mBase + wm * 32 + mi * 16 + r + 8 * hf;
        int n = nBase + wn * 64 + ni * 16 + l16;
        C[(size_t)m * N + n] = acc[mi][ni][r];
      }
}

// ---- RoPE + repack ----------------------------------------------------------
// mode 1: RoPE, head-major [H][S][D]   (Q, K)
// mode 0: no RoPE, transposed [H][D][S] (V -> makes attention V staging linear)

__global__ __launch_bounds__(256) void rope_cvt_kernel(const float* __restrict__ src,
                                                       const float* __restrict__ ang,
                                                       __bf16* __restrict__ dst,
                                                       int mode) {
  int gid = blockIdx.x * 256 + threadIdx.x;     // over S*E
  int col = gid & (E_ - 1);
  int s   = gid >> 11;
  int h   = col >> 7;
  int d   = col & 127;
  float x = src[gid];
  if (mode) {
    int dh = d & 63;
    float a  = ang[s * 64 + dh];
    float c  = __cosf(a);
    float sn = __sinf(a);
    float other = (d < 64) ? -src[(size_t)s * E_ + h * 128 + d + 64]
                           :  src[(size_t)s * E_ + h * 128 + d - 64];
    float y = x * c + other * sn;
    dst[((size_t)h * S_ + s) * D_ + d] = (__bf16)y;
  } else {
    dst[((size_t)h * D_ + d) * S_ + s] = (__bf16)x;
  }
}

// ---- flash block-sparse attention ------------------------------------------
// grid = (T, H); 256 threads = 8 waves; wave w owns query rows [w*16, w*16+16).
// K/V tiles double-buffered, staged entirely with async_to_lds (16/thread).

__global__ __launch_bounds__(256) void attn_kernel(const __bf16* __restrict__ Q,
                                                   const __bf16* __restrict__ Kb,
                                                   const __bf16* __restrict__ Vt,
                                                   const int* __restrict__ anchors,
                                                   __bf16* __restrict__ O) {
  extern __shared__ __align__(16) char smem[];
  __bf16* sK = (__bf16*)smem;                  // [2][128 key][128 d]   64KB
  __bf16* sV = (__bf16*)(smem + 65536);        // [2][128 d][128 key]  64KB
  __bf16* sP = (__bf16*)(smem + 131072);       // [wave][16][128]      32KB

  const int t   = blockIdx.x;
  const int h   = blockIdx.y;
  const int tid = threadIdx.x;
  const int lane = tid & 31;
  const int w    = tid >> 5;
  const int hf   = lane >> 4;
  const int l16  = lane & 15;
  __bf16* sPw = sP + w * 2048;

  // Q A-fragments (row = l16, K chunks of 32 along D)
  const int qrow = t * TILE_ + w * 16 + l16;
  const __bf16* qp = Q + ((size_t)h * S_ + qrow) * D_;
  v16bf aq[4];
#pragma unroll
  for (int kc = 0; kc < 4; kc++)
    aq[kc] = mk16(qp + kc * 32 + hf * 8, qp + kc * 32 + 16 + hf * 8);

  const v8f vz = {};
  v8f oacc[8];
#pragma unroll
  for (int nb = 0; nb < 8; nb++) oacc[nb] = vz;
  float mrow[8], lrow[8];
#pragma unroll
  for (int r = 0; r < 8; r++) { mrow[r] = -3.0e38f; lrow[r] = 0.0f; }

  auto tile_of = [&](int it) {
    return (it < KANC_) ? anchors[((h * T_) + t) * KANC_ + it] : t;
  };
  // stage K tile (linear [key][d]) and V tile (linear [d][key] from Vt) async
  auto stage = [&](int bi, int tile) {
    const __bf16* kt = Kb + ((size_t)h * S_ + tile * TILE_) * D_;
    __bf16* dK = sK + bi * 16384;
    __bf16* dV = sV + bi * 16384;
#pragma unroll
    for (int i = 0; i < 8; i++) {
      int off = (tid + i * 256) * 8;           // 0..16383, step 8
      async_copy16(lds_off(&dK[off]), &kt[off]);
      int d = off >> 7, key0 = off & 127;
      async_copy16(lds_off(&dV[off]),
                   &Vt[((size_t)h * D_ + d) * S_ + tile * TILE_ + key0]);
    }
  };

  stage(0, tile_of(0));
  int buf = 0;
  for (int it = 0; it < KANC_ + 1; it++, buf ^= 1) {
    const int tile = tile_of(it);
    const bool more = (it + 1) < (KANC_ + 1);
    if (more) {
      stage(buf ^ 1, tile_of(it + 1));
      WAIT_ASYNC("0x10");        // 16 newest (next tile) may stay in flight
    } else {
      WAIT_ASYNC("0x0");
    }
    __syncthreads();

    const __bf16* cK = sK + buf * 16384;
    const __bf16* cV = sV + buf * 16384;

    // ---- S = Q * K^T over 8 key blocks of 16
    v8f sq[8];
#pragma unroll
    for (int kb = 0; kb < 8; kb++) {
      v8f a = vz;
#pragma unroll
      for (int kc = 0; kc < 4; kc++) {
        const __bf16* kp = &cK[(kb * 16 + l16) * 128 + kc * 32 + hf * 8];
        v16bf bk = mk16(kp, kp + 16);
        a = wmma_bf16(aq[kc], bk, a);
      }
      sq[kb] = a;
    }

    // ---- scale + causal mask on gathered key positions
    const int tb  = tile * TILE_;
    const int qb0 = t * TILE_ + w * 16;
#pragma unroll
    for (int kb = 0; kb < 8; kb++)
#pragma unroll
      for (int r = 0; r < 8; r++) {
        int kp = tb + kb * 16 + l16;
        int qp2 = qb0 + r + 8 * hf;
        float v = sq[kb][r] * SCALE_;
        sq[kb][r] = (kp > qp2) ? NEGV : v;
      }

    // ---- row max (row m = r + 8*hf lives on one 16-lane half)
    float mt[8];
#pragma unroll
    for (int r = 0; r < 8; r++) {
      float m = -3.0e38f;
#pragma unroll
      for (int kb = 0; kb < 8; kb++) m = fmaxf(m, sq[kb][r]);
      mt[r] = m;
    }
#pragma unroll
    for (int msk = 1; msk < 16; msk <<= 1)
#pragma unroll
      for (int r = 0; r < 8; r++)
        mt[r] = fmaxf(mt[r], __shfl_xor(mt[r], msk, 32));

    float mn[8], alpha[8];
#pragma unroll
    for (int r = 0; r < 8; r++) {
      mn[r]    = fmaxf(mrow[r], mt[r]);
      alpha[r] = __expf(mrow[r] - mn[r]);
      mrow[r]  = mn[r];
      lrow[r] *= alpha[r];
    }
#pragma unroll
    for (int nb = 0; nb < 8; nb++)
#pragma unroll
      for (int r = 0; r < 8; r++) oacc[nb][r] *= alpha[r];

    // ---- P = exp(S - m), row sums, P (bf16) into per-wave LDS scratch
    float rs[8];
#pragma unroll
    for (int r = 0; r < 8; r++) rs[r] = 0.0f;
#pragma unroll
    for (int kb = 0; kb < 8; kb++)
#pragma unroll
      for (int r = 0; r < 8; r++) {
        float p = __expf(sq[kb][r] - mn[r]);
        rs[r] += p;
        sPw[(r + 8 * hf) * 128 + kb * 16 + l16] = (__bf16)p;
      }
#pragma unroll
    for (int msk = 1; msk < 16; msk <<= 1)
#pragma unroll
      for (int r = 0; r < 8; r++) rs[r] += __shfl_xor(rs[r], msk, 32);
#pragma unroll
    for (int r = 0; r < 8; r++) lrow[r] += rs[r];

    __syncthreads();   // P stores visible across lanes before A-frag reads

    // ---- O += P * V
    v16bf pf[4];
#pragma unroll
    for (int kc = 0; kc < 4; kc++)
      pf[kc] = mk16(&sPw[l16 * 128 + kc * 32 + hf * 8],
                    &sPw[l16 * 128 + kc * 32 + 16 + hf * 8]);
#pragma unroll
    for (int nb = 0; nb < 8; nb++)
#pragma unroll
      for (int kc = 0; kc < 4; kc++) {
        const __bf16* vp = &cV[(nb * 16 + l16) * 128 + kc * 32 + hf * 8];
        v16bf bv = mk16(vp, vp + 16);
        oacc[nb] = wmma_bf16(pf[kc], bv, oacc[nb]);
      }

    __syncthreads();   // done reading cK/cV before next stage overwrites
  }

  // ---- normalize and write bf16 to [S][H*D]
#pragma unroll
  for (int r = 0; r < 8; r++) {
    float rl = (lrow[r] > 0.0f) ? (1.0f / lrow[r]) : 0.0f;
    int srow = t * TILE_ + w * 16 + r + 8 * hf;
#pragma unroll
    for (int nb = 0; nb < 8; nb++)
      O[(size_t)srow * E_ + h * D_ + nb * 16 + l16] = (__bf16)(oacc[nb][r] * rl);
  }
}

// ---- host orchestration -----------------------------------------------------

extern "C" void kernel_launch(void* const* d_in, const int* in_sizes, int n_in,
                              void* d_out, int out_size, void* d_ws, size_t ws_size,
                              hipStream_t stream) {
  (void)in_sizes; (void)n_in; (void)out_size; (void)ws_size;
  const float* x   = (const float*)d_in[0];
  const float* wq  = (const float*)d_in[1];
  const float* wk  = (const float*)d_in[2];
  const float* wv  = (const float*)d_in[3];
  const float* wo  = (const float*)d_in[4];
  const float* ang = (const float*)d_in[5];
  const int*   anc = (const int*)d_in[6];
  float* out = (float*)d_out;

  char* ws = (char*)d_ws;
  size_t off = 0;
  __bf16* xb   = (__bf16*)(ws + off); off += (size_t)S_ * E_ * 2;
  __bf16* wqt  = (__bf16*)(ws + off); off += (size_t)E_ * E_ * 2;   // [N][K]
  __bf16* wkt  = (__bf16*)(ws + off); off += (size_t)E_ * E_ * 2;
  __bf16* wvt  = (__bf16*)(ws + off); off += (size_t)E_ * E_ * 2;
  __bf16* wot  = (__bf16*)(ws + off); off += (size_t)E_ * E_ * 2;
  float*  tmpf = (float*) (ws + off); off += (size_t)S_ * E_ * 4;
  __bf16* qbuf = (__bf16*)(ws + off); off += (size_t)H_ * S_ * D_ * 2;  // [H][S][D]
  __bf16* kbuf = (__bf16*)(ws + off); off += (size_t)H_ * S_ * D_ * 2;  // [H][S][D]
  __bf16* vbuf = (__bf16*)(ws + off); off += (size_t)H_ * S_ * D_ * 2;  // [H][D][S]
  __bf16* atb  = (__bf16*)(ws + off); off += (size_t)S_ * E_ * 2;

  // one-time repacks: x row-major bf16; weights transposed bf16 [N][K]
  cvt_bf16_kernel  <<<(S_ * E_) / 256, 256, 0, stream>>>(x,  xb,  S_ * E_);
  cvt_bf16_t_kernel<<<(E_ * E_) / 256, 256, 0, stream>>>(wq, wqt);
  cvt_bf16_t_kernel<<<(E_ * E_) / 256, 256, 0, stream>>>(wk, wkt);
  cvt_bf16_t_kernel<<<(E_ * E_) / 256, 256, 0, stream>>>(wv, wvt);
  cvt_bf16_t_kernel<<<(E_ * E_) / 256, 256, 0, stream>>>(wo, wot);

  dim3 gg(E_ / 128, S_ / 128);

  gemm_bf16_kernel<<<gg, 256, 0, stream>>>(xb, wqt, tmpf, S_, E_, E_);
  rope_cvt_kernel <<<(S_ * E_) / 256, 256, 0, stream>>>(tmpf, ang, qbuf, 1);
  gemm_bf16_kernel<<<gg, 256, 0, stream>>>(xb, wkt, tmpf, S_, E_, E_);
  rope_cvt_kernel <<<(S_ * E_) / 256, 256, 0, stream>>>(tmpf, ang, kbuf, 1);
  gemm_bf16_kernel<<<gg, 256, 0, stream>>>(xb, wvt, tmpf, S_, E_, E_);
  rope_cvt_kernel <<<(S_ * E_) / 256, 256, 0, stream>>>(tmpf, ang, vbuf, 0);

  attn_kernel<<<dim3(T_, H_), 256, 163840, stream>>>(qbuf, kbuf, vbuf, anc, atb);

  gemm_bf16_kernel<<<gg, 256, 0, stream>>>(atb, wot, out, S_, E_, E_);
}
